// CasualSelfAttention_79980880986744
// MI455X (gfx1250) — matrix operand order repaired and checked by reading, not compiled, not run on hardware
//
#include <hip/hip_runtime.h>

typedef __attribute__((ext_vector_type(16))) _Float16 v16h;
typedef __attribute__((ext_vector_type(8)))  _Float16 v8h;
typedef __attribute__((ext_vector_type(4)))  _Float16 v4h;
typedef __attribute__((ext_vector_type(8)))  float    v8f;

constexpr int Bc = 4;
constexpr int Tc = 2048;
constexpr int Cc = 1024;
constexpr int Hc = 16;
constexpr int Gc = 4;
constexpr int Dc = 64;
constexpr float NEG_INF = -1e30f;

#define WMMA_F16(a, b, c) \
  __builtin_amdgcn_wmma_f32_16x16x32_f16(false, (a), false, (b), (short)0, (c), false, false)

// ---------------------------------------------------------------------------
// f32 -> f16 conversion (n multiple of 1024; 256 thr x 4 elems / block)
// ---------------------------------------------------------------------------
__global__ __launch_bounds__(256) void f32_to_f16_kernel(
    const float* __restrict__ src, _Float16* __restrict__ dst, int n) {
  int i = (blockIdx.x * 256 + threadIdx.x) * 4;
  if (i + 3 < n) {
    float4 v = *(const float4*)(src + i);
    v4h h;
    h[0] = (_Float16)v.x; h[1] = (_Float16)v.y;
    h[2] = (_Float16)v.z; h[3] = (_Float16)v.w;
    *(v4h*)(dst + i) = h;
  }
}

__device__ inline v16h combine16(v8h lo, v8h hi) {
  v16h a;
#pragma unroll
  for (int e = 0; e < 8; ++e) { a[e] = lo[e]; a[e + 8] = hi[e]; }
  return a;
}

// ---------------------------------------------------------------------------
// Per-wave 32x64 WMMA GEMM tile, double-buffered k-loop.
// A fragment (f16 16x32): lane L (m=L&15, half=L>>4): halves e -> K=16*half+e
// (contiguous).  B fragment (f16 32x16): lane col=L&15,
// halves e -> K=16*half+e (contiguous).  B is [N,K] row-major (weights).
// ---------------------------------------------------------------------------
__device__ inline void wave_gemm_32x64(const _Float16* __restrict__ A, int lda,
                                       int row_base,
                                       const _Float16* __restrict__ B, int ldb,
                                       int n_base, int K, v8f acc[2][4]) {
  const int lane  = threadIdx.x & 31;
  const int m     = lane & 15;
  const int hi    = lane >> 4;
  const int abase = hi * 16;
  const int bbase = hi * 16;
  const _Float16* ar0 = A + (size_t)(row_base + m) * lda;
  const _Float16* ar1 = ar0 + (size_t)16 * lda;
  const _Float16* bc  = B + (size_t)(n_base + m) * ldb;

  // prefetch k0 = 0
  v8h a0l = *(const v8h*)(ar0 + abase);
  v8h a0h = *(const v8h*)(ar0 + 8 + abase);
  v8h a1l = *(const v8h*)(ar1 + abase);
  v8h a1h = *(const v8h*)(ar1 + 8 + abase);
  v16h bf[4];
#pragma unroll
  for (int nt = 0; nt < 4; ++nt)
    bf[nt] = *(const v16h*)(bc + (size_t)nt * 16 * ldb + bbase);

  for (int k0 = 0; k0 < K; k0 += 32) {
    const v16h a0 = combine16(a0l, a0h);
    const v16h a1 = combine16(a1l, a1h);
    v16h bcur[4];
#pragma unroll
    for (int nt = 0; nt < 4; ++nt) bcur[nt] = bf[nt];

    const int kn = k0 + 32;
    if (kn < K) {  // issue next-iteration loads before consuming current
      a0l = *(const v8h*)(ar0 + kn + abase);
      a0h = *(const v8h*)(ar0 + kn + 8 + abase);
      a1l = *(const v8h*)(ar1 + kn + abase);
      a1h = *(const v8h*)(ar1 + kn + 8 + abase);
#pragma unroll
      for (int nt = 0; nt < 4; ++nt)
        bf[nt] = *(const v16h*)(bc + (size_t)nt * 16 * ldb + kn + bbase);
    }
#pragma unroll
    for (int nt = 0; nt < 4; ++nt)
      acc[0][nt] = WMMA_F16(a0, bcur[nt], acc[0][nt]);
#pragma unroll
    for (int nt = 0; nt < 4; ++nt)
      acc[1][nt] = WMMA_F16(a1, bcur[nt], acc[1][nt]);
  }
}

// ---------------------------------------------------------------------------
// Fused QKV projection: xh[B*T,C] x {Wq^T | Wkv^T} ->
//   q16[B,H,T,D], k16[B,G,T,D], vT[B,G,D,T]  (V stored transposed!)
// grid: x=3 (8 waves * 64 cols), y=256 (32 rows per block)
// ---------------------------------------------------------------------------
__global__ __launch_bounds__(256) void qkv_gemm_kernel(
    const _Float16* __restrict__ xh,
    const _Float16* __restrict__ Wq16, const _Float16* __restrict__ Wkv16,
    _Float16* __restrict__ q16, _Float16* __restrict__ k16,
    _Float16* __restrict__ vT) {
  const int wave = threadIdx.x >> 5;
  const int lane = threadIdx.x & 31;
  const int col64 = blockIdx.x * 8 + wave;   // 0..23
  const int row_base = blockIdx.y * 32;
  const int n_base = col64 * 64;
  const bool is_q = (n_base < Cc);
  const _Float16* W = is_q ? Wq16 : Wkv16;
  const int nb = is_q ? n_base : n_base - Cc;

  v8f acc[2][4] = {};
  wave_gemm_32x64(xh, Cc, row_base, W, Cc, nb, Cc, acc);

  const int hi = lane >> 4;
#pragma unroll
  for (int ht = 0; ht < 2; ++ht) {
#pragma unroll
    for (int nt = 0; nt < 4; ++nt) {
      const int n = nb + nt * 16 + (lane & 15);
#pragma unroll
      for (int r = 0; r < 8; ++r) {
        const int M = row_base + ht * 16 + r + hi * 8;
        const int b = M >> 11;       // / T
        const int t = M & (Tc - 1);
        const _Float16 val = (_Float16)acc[ht][nt][r];
        if (is_q) {
          const int h = n >> 6, d = n & 63;
          q16[(((size_t)(b * Hc + h)) * Tc + t) * Dc + d] = val;
        } else if (n < Gc * Dc) {
          const int g = n >> 6, d = n & 63;
          k16[(((size_t)(b * Gc + g)) * Tc + t) * Dc + d] = val;
        } else {
          const int n2 = n - Gc * Dc;
          const int g = n2 >> 6, d = n2 & 63;
          vT[(((size_t)(b * Gc + g)) * Dc + d) * Tc + t] = val;  // [B,G,D,T]
        }
      }
    }
  }
}

// ---------------------------------------------------------------------------
// Flash attention fragment loads
// ---------------------------------------------------------------------------
__device__ inline void load_kfrag(const _Float16* __restrict__ kp, int kb,
                                  int lane, v16h out[4]) {
  const int bbase = (lane >> 4) * 16;
#pragma unroll
  for (int st = 0; st < 2; ++st) {
    const _Float16* krow = kp + (size_t)(kb + st * 16 + (lane & 15)) * Dc;
    out[st * 2 + 0] = *(const v16h*)(krow + bbase);
    out[st * 2 + 1] = *(const v16h*)(krow + 32 + bbase);
  }
}

__device__ inline void load_vfrag(const _Float16* __restrict__ vpT, int kb,
                                  int lane, v16h out[4]) {
  const int bbase = (lane >> 4) * 16;
#pragma unroll
  for (int nt = 0; nt < 4; ++nt) {
    const int d = nt * 16 + (lane & 15);
    out[nt] = *(const v16h*)(vpT + (size_t)d * Tc + kb + bbase);
  }
}

// ---------------------------------------------------------------------------
// Flash attention: one wave per (b, h, 16-query tile). Online softmax over
// 32-key chunks; S = Q*K^T via 2x2 WMMA, O += P*V via 4 WMMA.
// Next chunk's K/V fragments prefetched under the softmax VALU chain.
// ---------------------------------------------------------------------------
__global__ __launch_bounds__(256) void flash_attn_kernel(
    const _Float16* __restrict__ q16, const _Float16* __restrict__ k16,
    const _Float16* __restrict__ vT, _Float16* __restrict__ y16) {
  __shared__ __align__(32) _Float16 plds_all[8][16 * 32];
  const int wave = threadIdx.x >> 5;
  const int lane = threadIdx.x & 31;
  _Float16* plds = plds_all[wave];

  const int lin = blockIdx.x * 8 + wave;     // 0..8191
  const int qt  = lin & 127;
  const int h   = (lin >> 7) & 15;
  const int b   = lin >> 11;
  const int g   = h & (Gc - 1);              // head h uses kv group h % G
  const int qbase = qt * 16;

  const int m = lane & 15, hi = lane >> 4;
  const int abase = hi * 16;
  const float scale = 0.125f;                // 1/sqrt(64)

  // Q tile in A-layout, pre-scaled (vector loads)
  v16h qa[2];
  const _Float16* qrow =
      q16 + (((size_t)(b * Hc + h)) * Tc + qbase + m) * Dc;
#pragma unroll
  for (int c = 0; c < 2; ++c) {
    v8h lo = *(const v8h*)(qrow + 32 * c + abase);
    v8h hh = *(const v8h*)(qrow + 32 * c + 8 + abase);
#pragma unroll
    for (int e = 0; e < 8; ++e) {
      qa[c][e]     = (_Float16)((float)lo[e] * scale);
      qa[c][e + 8] = (_Float16)((float)hh[e] * scale);
    }
  }

  v8f o[4] = {};
  float rmax[8], rsum[8];
#pragma unroll
  for (int r = 0; r < 8; ++r) { rmax[r] = -INFINITY; rsum[r] = 0.0f; }

  const _Float16* kp  = k16 + ((size_t)(b * Gc + g)) * Tc * Dc;
  const _Float16* vpT = vT  + ((size_t)(b * Gc + g)) * Dc * Tc;

  const int nchunks = (qbase + 16 + 31) >> 5;

  v16h kf[4], vf[4];
  load_kfrag(kp, 0, lane, kf);
  load_vfrag(vpT, 0, lane, vf);

  for (int kc = 0; kc < nchunks; ++kc) {
    const int kb = kc * 32;
    // S = Q * K^T : two 16x16 score tiles
    const v8f vzero = {};
    v8f s[2];
    s[0] = vzero; s[1] = vzero;
#pragma unroll
    for (int st = 0; st < 2; ++st) {
      s[st] = WMMA_F16(qa[0], kf[st * 2 + 0], s[st]);
      s[st] = WMMA_F16(qa[1], kf[st * 2 + 1], s[st]);
    }
    // keep current V, prefetch next chunk's K/V under the softmax chain
    v16h vcur[4];
#pragma unroll
    for (int i = 0; i < 4; ++i) vcur[i] = vf[i];
    if (kc + 1 < nchunks) {
      load_kfrag(kp, kb + 32, lane, kf);
      load_vfrag(vpT, kb + 32, lane, vf);
    }

    // causal mask + online softmax (C-layout: row = r+8*hi, col = lane&15)
    float alpha[8];
#pragma unroll
    for (int r = 0; r < 8; ++r) {
      const int qi = qbase + r + hi * 8;
      const int j0 = kb + (lane & 15);
      float s0 = (j0      <= qi) ? s[0][r] : NEG_INF;
      float s1 = (j0 + 16 <= qi) ? s[1][r] : NEG_INF;
      float mx = fmaxf(s0, s1);
      mx = fmaxf(mx, __shfl_xor(mx, 1));
      mx = fmaxf(mx, __shfl_xor(mx, 2));
      mx = fmaxf(mx, __shfl_xor(mx, 4));
      mx = fmaxf(mx, __shfl_xor(mx, 8));
      const float newmax = fmaxf(rmax[r], mx);
      alpha[r] = __expf(rmax[r] - newmax);
      rmax[r] = newmax;
      const float p0 = __expf(s0 - newmax);
      const float p1 = __expf(s1 - newmax);
      float ps = p0 + p1;
      ps += __shfl_xor(ps, 1);
      ps += __shfl_xor(ps, 2);
      ps += __shfl_xor(ps, 4);
      ps += __shfl_xor(ps, 8);
      rsum[r] = rsum[r] * alpha[r] + ps;
      plds[(r + hi * 8) * 32 + (lane & 15)]      = (_Float16)p0;
      plds[(r + hi * 8) * 32 + 16 + (lane & 15)] = (_Float16)p1;
    }
#pragma unroll
    for (int nt = 0; nt < 4; ++nt)
#pragma unroll
      for (int r = 0; r < 8; ++r) o[nt][r] *= alpha[r];

    // P back in A-layout (vector LDS reads)
    v8h plo = *(const v8h*)(plds + m * 32 + abase);
    v8h phi = *(const v8h*)(plds + m * 32 + 8 + abase);
    const v16h pa = combine16(plo, phi);

    // O += P * V   (V fragment from transposed [D,T] layout, vector loads)
#pragma unroll
    for (int nt = 0; nt < 4; ++nt)
      o[nt] = WMMA_F16(pa, vcur[nt], o[nt]);
  }

  // normalize and store y16[b, t, h*D + d]
  float rinv[8];
#pragma unroll
  for (int r = 0; r < 8; ++r) rinv[r] = 1.0f / rsum[r];
#pragma unroll
  for (int nt = 0; nt < 4; ++nt) {
    const int d = nt * 16 + (lane & 15);
#pragma unroll
    for (int r = 0; r < 8; ++r) {
      const int t = qbase + r + hi * 8;
      y16[((size_t)b * Tc + t) * Cc + h * Dc + d] =
          (_Float16)(o[nt][r] * rinv[r]);
    }
  }
}

// ---------------------------------------------------------------------------
// Output projection: out = y16 @ Wproj^T + b  (f32 out)
// grid: x=2, y=256
// ---------------------------------------------------------------------------
__global__ __launch_bounds__(256) void proj_gemm_kernel(
    const _Float16* __restrict__ y16, const _Float16* __restrict__ Wp16,
    const float* __restrict__ bias, float* __restrict__ out) {
  const int wave = threadIdx.x >> 5;
  const int lane = threadIdx.x & 31;
  const int col64 = blockIdx.x * 8 + wave;   // 0..15
  const int row_base = blockIdx.y * 32;
  const int n_base = col64 * 64;

  v8f acc[2][4] = {};
  wave_gemm_32x64(y16, Cc, row_base, Wp16, Cc, n_base, Cc, acc);

  const int hi = lane >> 4;
#pragma unroll
  for (int ht = 0; ht < 2; ++ht) {
#pragma unroll
    for (int nt = 0; nt < 4; ++nt) {
      const int n = n_base + nt * 16 + (lane & 15);
      const float bv = bias[n];
#pragma unroll
      for (int r = 0; r < 8; ++r) {
        const int M = row_base + ht * 16 + r + hi * 8;
        out[(size_t)M * Cc + n] = acc[ht][nt][r] + bv;
      }
    }
  }
}

// ---------------------------------------------------------------------------
extern "C" void kernel_launch(void* const* d_in, const int* in_sizes, int n_in,
                              void* d_out, int out_size, void* d_ws,
                              size_t ws_size, hipStream_t stream) {
  const float* x     = (const float*)d_in[0];
  const float* Wq    = (const float*)d_in[1];
  const float* Wkv   = (const float*)d_in[2];
  const float* Wproj = (const float*)d_in[3];
  const float* bproj = (const float*)d_in[4];
  float* out = (float*)d_out;

  char* ws = (char*)d_ws;
  const size_t n_x   = (size_t)Bc * Tc * Cc;          // 8388608
  const size_t n_wq  = (size_t)Cc * Cc;               // 1048576
  const size_t n_wkv = (size_t)2 * Gc * Dc * Cc;      // 524288
  const size_t n_q   = (size_t)Bc * Hc * Tc * Dc;     // 8388608
  const size_t n_kv  = (size_t)Bc * Gc * Tc * Dc;     // 2097152

  _Float16* xh   = (_Float16*)ws;  ws += n_x * 2;
  _Float16* Wq16 = (_Float16*)ws;  ws += n_wq * 2;
  _Float16* Wk16 = (_Float16*)ws;  ws += n_wkv * 2;
  _Float16* Wp16 = (_Float16*)ws;  ws += n_wq * 2;
  _Float16* q16  = (_Float16*)ws;  ws += n_q * 2;
  _Float16* k16  = (_Float16*)ws;  ws += n_kv * 2;
  _Float16* vTw  = (_Float16*)ws;  ws += n_kv * 2;
  _Float16* y16  = (_Float16*)ws;  ws += n_x * 2;

  f32_to_f16_kernel<<<(int)(n_x   / 1024), 256, 0, stream>>>(x, xh, (int)n_x);
  f32_to_f16_kernel<<<(int)(n_wq  / 1024), 256, 0, stream>>>(Wq, Wq16, (int)n_wq);
  f32_to_f16_kernel<<<(int)(n_wkv / 1024), 256, 0, stream>>>(Wkv, Wk16, (int)n_wkv);
  f32_to_f16_kernel<<<(int)(n_wq  / 1024), 256, 0, stream>>>(Wproj, Wp16, (int)n_wq);

  dim3 g1(3, 256);
  qkv_gemm_kernel<<<g1, 256, 0, stream>>>(xh, Wq16, Wk16, q16, k16, vTw);

  flash_attn_kernel<<<1024, 256, 0, stream>>>(q16, k16, vTw, y16);

  dim3 g2(2, 256);
  proj_gemm_kernel<<<g2, 256, 0, stream>>>(y16, Wp16, bproj, out);
}
